// SelfAttention_10617159156326
// MI455X (gfx1250) — matrix operand order repaired and checked by reading, not compiled, not run on hardware
//
#include <hip/hip_runtime.h>
#include <hip/hip_bf16.h>

// ---------------------------------------------------------------------------
// Self-attention layer for MI455X (gfx1250), wave32 + WMMA bf16.
//   phase 0: one-shot fp32 -> bf16 conversion of x, W_qkv, W_out
//   phase 1: qkv = x @ W_qkv^T + b_qkv   (bf16 WMMA GEMM, async-LDS staged,
//            scatter q / k / v^T bf16 into workspace)
//   phase 2: causal flash attention, online softmax, all-WMMA, bf16 out
//   phase 3: out = attn @ W_out^T + b_out (same GEMM, fp32 result)
// ---------------------------------------------------------------------------

typedef __attribute__((ext_vector_type(16))) __bf16 v16bf;
typedef __attribute__((ext_vector_type(8)))  __bf16 v8bf;
typedef __attribute__((ext_vector_type(8)))  float  v8f;
typedef __attribute__((ext_vector_type(4)))  float  v4f;
typedef __attribute__((ext_vector_type(4)))  int    v4i;

#define D_MODEL   2048
#define N_HEADS   16
#define HEAD_DIM  128
#define SEQ_LEN   2048
#define BATCH     2
#define M_TOTAL   (BATCH * SEQ_LEN)       // 4096

#define LDSR 40   // bf16 elems per LDS row (80B: 16B-aligned, skews banks)

static __device__ inline v16bf cat16(v8bf lo, v8bf hi) {
  return __builtin_shufflevector(lo, hi, 0,1,2,3,4,5,6,7,8,9,10,11,12,13,14,15);
}

static __device__ inline v8bf pack8(v4f a, v4f b) {
  v8bf r;
  r[0] = (__bf16)a[0]; r[1] = (__bf16)a[1]; r[2] = (__bf16)a[2]; r[3] = (__bf16)a[3];
  r[4] = (__bf16)b[0]; r[5] = (__bf16)b[1]; r[6] = (__bf16)b[2]; r[7] = (__bf16)b[3];
  return r;
}

static __device__ inline v8f wmma_bf16(v16bf a, v16bf b, v8f c) {
  return __builtin_amdgcn_wmma_f32_16x16x32_bf16(
      /*neg_a=*/false, a, /*neg_b=*/false, b,
      /*c_mod=*/(short)0, c, /*reuse_a=*/false, /*reuse_b=*/false);
}

// 16-byte global -> LDS copy via the gfx1250 async path (ASYNCcnt-tracked,
// bypasses VGPRs). Builtin signature: (v4i AS1* gsrc, v4i AS3* ldst, Ii, Ii).
static __device__ inline void async_cp16(const __bf16* g, __bf16* l) {
#if defined(__has_builtin) && __has_builtin(__builtin_amdgcn_global_load_async_to_lds_b128)
  __builtin_amdgcn_global_load_async_to_lds_b128(
      (__attribute__((address_space(1))) v4i*)(unsigned long long)(uintptr_t)g,
      (__attribute__((address_space(3))) v4i*)(unsigned)(uintptr_t)l,
      0, 0);
#else
  *(v8bf*)l = *(const v8bf*)g;
#endif
}

static __device__ inline void wait_async_all() {
  asm volatile("s_wait_asynccnt 0x0" ::: "memory");
}

// ---------------------------------------------------------------------------
// Phase 0: fp32 -> bf16 elementwise convert (n8 = n/8, n divisible by 8)
// ---------------------------------------------------------------------------
__global__ __launch_bounds__(256)
void cvt_bf16(const float* __restrict__ src, __bf16* __restrict__ dst, int n8) {
  const int i = blockIdx.x * 256 + threadIdx.x;
  if (i < n8) {
    v4f a = *(const v4f*)(src + (size_t)i * 8);
    v4f b = *(const v4f*)(src + (size_t)i * 8 + 4);
    *(v8bf*)(dst + (size_t)i * 8) = pack8(a, b);
  }
}

// ---------------------------------------------------------------------------
// GEMM: C[M,N] = A[M,K](bf16) * B[N,K](bf16) + bias(f32)
// Block: 256 thr (8 waves), tile 128x128, BK=32, double-buffered async LDS.
// Wave tile 32x64 (2x4 WMMA accumulators).
// MODE 0: scatter bf16 q / k / v^T into attention workspace
// MODE 1: fp32 result
// ---------------------------------------------------------------------------
template <int MODE>
__global__ __launch_bounds__(256)
void gemm_wmma(const __bf16* __restrict__ A, const __bf16* __restrict__ Bw,
               const float* __restrict__ bias,
               __bf16* __restrict__ q_ws, __bf16* __restrict__ k_ws,
               __bf16* __restrict__ v_ws, float* __restrict__ out,
               int M, int N, int K) {
  __shared__ __bf16 sA[2][128 * LDSR];   // 2 x 10 KB
  __shared__ __bf16 sB[2][128 * LDSR];   // 2 x 10 KB

  const int t    = threadIdx.x;
  const int wave = t >> 5;
  const int lane = t & 31;
  const int half = lane >> 4;
  const int l16  = lane & 15;
  const int wm   = wave >> 1;       // 0..3 : 32-row strip
  const int wn   = wave & 1;        // 0..1 : 64-col strip
  const int bm   = blockIdx.y * 128;
  const int bn   = blockIdx.x * 128;

  // copy assignment: thread t stages 2x16B of A and of B per tile
  const int crow = t >> 1;          // 0..127
  const int ccol = (t & 1) * 16;    // 0 / 16
  const __bf16* gA = A  + (size_t)(bm + crow) * K + ccol;
  const __bf16* gB = Bw + (size_t)(bn + crow) * K + ccol;

  v8f acc[2][4] = {};

  // prime buffer 0
  {
    __bf16* la = &sA[0][crow * LDSR + ccol];
    __bf16* lb = &sB[0][crow * LDSR + ccol];
    async_cp16(gA, la);     async_cp16(gA + 8, la + 8);
    async_cp16(gB, lb);     async_cp16(gB + 8, lb + 8);
  }
  wait_async_all();
  __syncthreads();

  int buf = 0;
  for (int k0 = 0; k0 < K; k0 += 32) {
    // kick off next tile into the other buffer (overlaps with WMMAs)
    if (k0 + 32 < K) {
      const __bf16* pa = gA + k0 + 32;
      const __bf16* pb = gB + k0 + 32;
      __bf16* la = &sA[buf ^ 1][crow * LDSR + ccol];
      __bf16* lb = &sB[buf ^ 1][crow * LDSR + ccol];
      async_cp16(pa, la);   async_cp16(pa + 8, la + 8);
      async_cp16(pb, lb);   async_cp16(pb + 8, lb + 8);
    }

    // A fragments: row = lane%16, elems: K = (e/8)*16 + half*8 + e%8
    v16bf af[2];
#pragma unroll
    for (int mi = 0; mi < 2; ++mi) {
      const __bf16* p = &sA[buf][(wm * 32 + mi * 16 + l16) * LDSR + half * 8];
      af[mi] = cat16(*(const v8bf*)p, *(const v8bf*)(p + 16));
    }
    // B fragments: col = lane%16, elems: K = (e/8)*16 + half*8 + e%8
    v16bf bf[4];
#pragma unroll
    for (int ni = 0; ni < 4; ++ni) {
      const __bf16* p = &sB[buf][(wn * 64 + ni * 16 + l16) * LDSR + half * 8];
      bf[ni] = cat16(*(const v8bf*)p, *(const v8bf*)(p + 16));
    }
#pragma unroll
    for (int mi = 0; mi < 2; ++mi)
#pragma unroll
      for (int ni = 0; ni < 4; ++ni)
        acc[mi][ni] = wmma_bf16(af[mi], bf[ni], acc[mi][ni]);

    wait_async_all();     // our next-tile copies have landed
    __syncthreads();      // everyone's have; also fences readers of buf
    buf ^= 1;
  }

  // Epilogue. C layout: row m = r + half*8 (+tile), col n = lane%16 (+tile).
#pragma unroll
  for (int mi = 0; mi < 2; ++mi) {
#pragma unroll
    for (int ni = 0; ni < 4; ++ni) {
      const int n  = bn + wn * 64 + ni * 16 + l16;
      const float bv = bias[n];
#pragma unroll
      for (int r = 0; r < 8; ++r) {
        const int m = bm + wm * 32 + mi * 16 + r + half * 8;
        const float v = acc[mi][ni][r] + bv;
        if (MODE == 0) {
          const int bi   = m >> 11;            // batch
          const int row  = m & 2047;           // seq pos
          const int sel  = n >> 11;            // 0=q 1=k 2=v
          const int dd   = n & 2047;
          const int hh   = dd >> 7;
          const int feat = dd & 127;
          const size_t bh = (size_t)(bi * N_HEADS + hh);
          if (sel == 0)
            q_ws[(bh * SEQ_LEN + row) * HEAD_DIM + feat] = (__bf16)v;
          else if (sel == 1)
            k_ws[(bh * SEQ_LEN + row) * HEAD_DIM + feat] = (__bf16)v;
          else  // V stored transposed: [bh][feat][key]
            v_ws[(bh * HEAD_DIM + feat) * SEQ_LEN + row] = (__bf16)v;
        } else {
          out[(size_t)m * N + n] = v;
        }
      }
    }
  }
}

// ---------------------------------------------------------------------------
// Flash attention, causal. One wave per 16-query tile of one (batch, head).
// Q: 4 resident A-fragments (e=128). Per 32-key tile:
//   S = Q K^T : 8 WMMAs;  O += P V : 8 WMMAs.
// P routed C-frag -> A-frag through per-wave LDS (16x32 bf16).
// Output written as bf16 (feeds the bf16 out-projection GEMM directly).
// ---------------------------------------------------------------------------
__global__ __launch_bounds__(256)
void flash_attn(const __bf16* __restrict__ q_ws, const __bf16* __restrict__ k_ws,
                const __bf16* __restrict__ v_ws, __bf16* __restrict__ attn_out) {
  __shared__ __bf16 sP[8 * 16 * 32];

  const int wave = threadIdx.x >> 5;
  const int lane = threadIdx.x & 31;
  const int half = lane >> 4;
  const int l16  = lane & 15;

  const int gw = blockIdx.x * 8 + wave;        // 0..4095
  const int QT = SEQ_LEN / 16;                 // 128 query tiles per (b,h)
  const int qt = gw % QT;
  const int bh = gw / QT;                      // 0..31

  const size_t base = (size_t)bh * SEQ_LEN * HEAD_DIM;
  const __bf16* Q  = q_ws + base;
  const __bf16* Kp = k_ws + base;
  const __bf16* Vt = v_ws + base;              // [feat][key]

  v16bf qf[4];
  {
    const __bf16* qr = Q + (size_t)(qt * 16 + l16) * HEAD_DIM;
#pragma unroll
    for (int c = 0; c < 4; ++c) {
      const __bf16* p = qr + c * 32 + half * 8;
      qf[c] = cat16(*(const v8bf*)p, *(const v8bf*)(p + 16));
    }
  }

  v8f oacc[8] = {};
  float rmax[8], rsum[8];
#pragma unroll
  for (int r = 0; r < 8; ++r) { rmax[r] = -1e30f; rsum[r] = 0.f; }

  const float scale = 0.08838834764831845f;    // 1/sqrt(128)
  __bf16* pt = &sP[wave * 16 * 32];

  const int nkt = (qt * 16 + 16 + 31) >> 5;    // causal key-tile count
  for (int kt = 0; kt < nkt; ++kt) {
    const int kbase = kt * 32;

    v8f s[2] = {};
#pragma unroll
    for (int kh = 0; kh < 2; ++kh) {
      const __bf16* kr = Kp + (size_t)(kbase + kh * 16 + l16) * HEAD_DIM;
#pragma unroll
      for (int c = 0; c < 4; ++c) {
        const __bf16* p = kr + c * 32 + half * 8;
        v16bf kf = cat16(*(const v8bf*)p, *(const v8bf*)(p + 16));
        s[kh] = wmma_bf16(qf[c], kf, s[kh]);
      }
    }

#pragma unroll
    for (int kh = 0; kh < 2; ++kh) {
      const int kg = kbase + kh * 16 + l16;
#pragma unroll
      for (int r = 0; r < 8; ++r) {
        const int qg = qt * 16 + r + half * 8;
        float v = s[kh][r] * scale;
        s[kh][r] = (kg > qg) ? -1e30f : v;
      }
    }

    float tmax[8];
#pragma unroll
    for (int r = 0; r < 8; ++r) tmax[r] = fmaxf(s[0][r], s[1][r]);
#pragma unroll
    for (int msk = 1; msk < 16; msk <<= 1)
#pragma unroll
      for (int r = 0; r < 8; ++r)
        tmax[r] = fmaxf(tmax[r], __shfl_xor(tmax[r], msk, 32));

    float al[8];
#pragma unroll
    for (int r = 0; r < 8; ++r) {
      const float mn = fmaxf(rmax[r], tmax[r]);
      al[r] = __expf(rmax[r] - mn);
      rmax[r] = mn;
    }

    float ps[8];
#pragma unroll
    for (int r = 0; r < 8; ++r) ps[r] = 0.f;
#pragma unroll
    for (int kh = 0; kh < 2; ++kh)
#pragma unroll
      for (int r = 0; r < 8; ++r) {
        const float p = __expf(s[kh][r] - rmax[r]);
        s[kh][r] = p;
        ps[r] += p;
      }
#pragma unroll
    for (int msk = 1; msk < 16; msk <<= 1)
#pragma unroll
      for (int r = 0; r < 8; ++r) ps[r] += __shfl_xor(ps[r], msk, 32);
#pragma unroll
    for (int r = 0; r < 8; ++r) rsum[r] = rsum[r] * al[r] + ps[r];
#pragma unroll
    for (int ec = 0; ec < 8; ++ec)
#pragma unroll
      for (int r = 0; r < 8; ++r) oacc[ec][r] *= al[r];

    // P: C-frag -> LDS -> A-frag (same-wave LDS ops are in-order)
#pragma unroll
    for (int kh = 0; kh < 2; ++kh)
#pragma unroll
      for (int r = 0; r < 8; ++r)
        pt[(r + half * 8) * 32 + kh * 16 + l16] = (__bf16)s[kh][r];
    asm volatile("s_wait_dscnt 0" ::: "memory");
    const __bf16* pp = pt + l16 * 32 + half * 8;
    v16bf pf = cat16(*(const v8bf*)pp, *(const v8bf*)(pp + 16));

#pragma unroll
    for (int ec = 0; ec < 8; ++ec) {
      const __bf16* vp = Vt + (size_t)(ec * 16 + l16) * SEQ_LEN + kbase + half * 8;
      v16bf vf = cat16(*(const v8bf*)vp, *(const v8bf*)(vp + 16));
      oacc[ec] = wmma_bf16(pf, vf, oacc[ec]);
    }
  }

  // writeback: bf16 [b*l][d] with head-merged columns
  const int bi = bh >> 4, hh = bh & 15;
#pragma unroll
  for (int ec = 0; ec < 8; ++ec) {
#pragma unroll
    for (int r = 0; r < 8; ++r) {
      const int qg = qt * 16 + r + half * 8;
      const int feat = ec * 16 + l16;
      attn_out[((size_t)(bi * SEQ_LEN + qg)) * D_MODEL + hh * HEAD_DIM + feat] =
          (__bf16)(oacc[ec][r] / rsum[r]);
    }
  }
}

// ---------------------------------------------------------------------------
extern "C" void kernel_launch(void* const* d_in, const int* in_sizes, int n_in,
                              void* d_out, int out_size, void* d_ws, size_t ws_size,
                              hipStream_t stream) {
  const float* x     = (const float*)d_in[0];
  const float* W_qkv = (const float*)d_in[1];
  const float* b_qkv = (const float*)d_in[2];
  const float* W_out = (const float*)d_in[3];
  const float* b_out = (const float*)d_in[4];
  float* out = (float*)d_out;

  char* ws = (char*)d_ws;
  const size_t MB = 1048576;
  __bf16* xb   = (__bf16*)(ws);                 // 16 MB  x bf16
  __bf16* wqb  = (__bf16*)(ws + 16 * MB);       // 24 MB  W_qkv bf16
  __bf16* wob  = (__bf16*)(ws + 40 * MB);       //  8 MB  W_out bf16
  __bf16* q_ws = (__bf16*)(ws + 48 * MB);       // 16 MB
  __bf16* k_ws = (__bf16*)(ws + 64 * MB);       // 16 MB
  __bf16* v_ws = (__bf16*)(ws + 80 * MB);       // 16 MB (transposed V)
  __bf16* attn = (__bf16*)(ws + 96 * MB);       // 16 MB bf16 attn output

  // Phase 0: one-shot bf16 conversion
  cvt_bf16<<<(M_TOTAL * D_MODEL / 8 + 255) / 256, 256, 0, stream>>>(x, xb, M_TOTAL * D_MODEL / 8);
  cvt_bf16<<<(3 * D_MODEL * D_MODEL / 8 + 255) / 256, 256, 0, stream>>>(W_qkv, wqb, 3 * D_MODEL * D_MODEL / 8);
  cvt_bf16<<<(D_MODEL * D_MODEL / 8 + 255) / 256, 256, 0, stream>>>(W_out, wob, D_MODEL * D_MODEL / 8);

  // Phase 1: qkv projection  M=4096, N=6144, K=2048
  gemm_wmma<0><<<dim3(3 * D_MODEL / 128, M_TOTAL / 128), 256, 0, stream>>>(
      xb, wqb, b_qkv, q_ws, k_ws, v_ws, nullptr, M_TOTAL, 3 * D_MODEL, D_MODEL);

  // Phase 2: causal flash attention (4096 wave-tiles / 8 waves per block)
  flash_attn<<<512, 256, 0, stream>>>(q_ws, k_ws, v_ws, attn);

  // Phase 3: output projection  M=4096, N=2048, K=2048
  gemm_wmma<1><<<dim3(D_MODEL / 128, M_TOTAL / 128), 256, 0, stream>>>(
      attn, wob, b_out, nullptr, nullptr, nullptr, out, M_TOTAL, D_MODEL, D_MODEL);
}
